// SelfAttention_49280454754662
// MI455X (gfx1250) — matrix operand order repaired and checked by reading, not compiled, not run on hardware
//
#include <hip/hip_runtime.h>
#include <hip/hip_bf16.h>
#include <stdint.h>

// ---------------------------------------------------------------------------
// SelfAttention for MI455X (gfx1250): f16 WMMA everywhere, f32 accumulate.
// B=2, N=4096, D=512, H=8, Dh=64.
// Round 2: 32x32 register tiles in all GEMMs (2x fragment reuse), 32-row
// query tiles in attention (all 8 waves active in every phase, 256 KB LDS).
// ---------------------------------------------------------------------------

typedef _Float16 half_t;
typedef __attribute__((ext_vector_type(16))) _Float16 v16h;
typedef __attribute__((ext_vector_type(8)))  _Float16 v8h;
typedef __attribute__((ext_vector_type(8)))  float    v8f;

#define HEADS   8
#define NSEQ    4096
#define DMODEL  512
#define DHEAD   64
#define BATCH   2
#define MROWS   (BATCH * NSEQ)   // 8192

// D = A(16x32 f16) x B(32x16 f16) + C(16x16 f32)
__device__ __forceinline__ v8f wmma16(v16h a, v16h b, v8f c) {
  return __builtin_amdgcn_wmma_f32_16x16x32_f16(
      /*neg_a=*/false, a, /*neg_b=*/false, b,
      /*c_mod=*/(short)0, c, /*reuse_a=*/false, /*reuse_b=*/false);
}

// A fragment (16 rows x 32 K) from row-major f16, ld in elements.
// Layout: lane<16 -> row=lane, K=0..15; lane>=16 -> K=16..31 (matches load_B).
__device__ __forceinline__ v16h load_A(const half_t* base, int ld, int lane) {
  int r  = lane & 15;
  int ko = (lane >> 4) << 4;                 // 0 or 16
  const half_t* p = base + (size_t)r * ld + ko;
  v8h lo = *(const v8h*)p;                   // K = ko .. ko+7
  v8h hi = *(const v8h*)(p + 8);             // K = ko+8 .. ko+15
  v16h a;
#pragma unroll
  for (int i = 0; i < 8; ++i) { a[i] = lo[i]; a[i + 8] = hi[i]; }
  return a;
}

// B fragment (32 K x 16 cols) from "Bt" storage: row n = output column,
// contiguous over K, ld in elements. Lane: col=lane&15, K range = (lane>>4)*16.
__device__ __forceinline__ v16h load_B(const half_t* base, int ld, int lane) {
  int col = lane & 15;
  int ks  = (lane >> 4) << 4;                // 0 or 16
  return *(const v16h*)(base + (size_t)col * ld + ks);
}

__device__ __forceinline__ float wave_sum(float v) {
#pragma unroll
  for (int o = 16; o > 0; o >>= 1) v += __shfl_xor(v, o, 32);
  return v;
}
__device__ __forceinline__ float wave_max(float v) {
#pragma unroll
  for (int o = 16; o > 0; o >>= 1) v = fmaxf(v, __shfl_xor(v, o, 32));
  return v;
}

// ---------------------------------------------------------------------------
// W (K x Ncols f32, row-major) -> Wt (Ncols x K f16, row-major)
__global__ void wt_kernel(const float* __restrict__ W, half_t* __restrict__ Wt,
                          int K, int Ncols) {
  size_t tid = (size_t)blockIdx.x * blockDim.x + threadIdx.x;
  if (tid >= (size_t)K * Ncols) return;
  int n = (int)(tid % Ncols);
  int k = (int)(tid / Ncols);
  Wt[(size_t)n * K + k] = (half_t)W[tid];
}

// ---------------------------------------------------------------------------
// LayerNorm over 512-wide rows, one wave per row, f16 output.
__global__ void __launch_bounds__(256) ln_kernel(const float* __restrict__ x,
    const float* __restrict__ gamma, const float* __restrict__ beta,
    half_t* __restrict__ y) {
  int lane = threadIdx.x & 31;
  int wave = threadIdx.x >> 5;
  size_t row = (size_t)blockIdx.x * 8 + wave;
  const float* xr = x + row * DMODEL;
  float r[16];
  float s = 0.f;
#pragma unroll
  for (int t = 0; t < 16; ++t) { r[t] = xr[lane + 32 * t]; s += r[t]; }
  float mu = wave_sum(s) * (1.0f / DMODEL);
  float v = 0.f;
#pragma unroll
  for (int t = 0; t < 16; ++t) { float d = r[t] - mu; v += d * d; }
  float inv = rsqrtf(wave_sum(v) * (1.0f / DMODEL) + 1e-5f);
  half_t* yr = y + row * DMODEL;
#pragma unroll
  for (int t = 0; t < 16; ++t) {
    int j = lane + 32 * t;
    yr[j] = (half_t)((r[t] - mu) * inv * gamma[j] + beta[j]);
  }
}

// ---------------------------------------------------------------------------
// Projection GEMM: Y = X(8192x512 f16) @ Wt^T, per-wave 32x32 tile, K=512.
// do_clip: clamp +-5 (Q/K). transpose_out: write [b,h,64,N] (V) else [b,h,N,64].
__global__ void __launch_bounds__(256) proj_kernel(const half_t* __restrict__ X,
    const half_t* __restrict__ Wt, half_t* __restrict__ out,
    int do_clip, int transpose_out) {
  int lane = threadIdx.x & 31;
  int wave = threadIdx.x >> 5;
  int gw = blockIdx.x * 8 + wave;            // 4096 wave-tiles (32x32)
  int mt = gw >> 4;                          // 256 row tiles of 32
  int ct = gw & 15;                          // 16 col tiles of 32
  const half_t* ap = X + (size_t)mt * 32 * DMODEL;
  const half_t* bp = Wt + (size_t)ct * 32 * DMODEL;
  v8f c00 = {}, c01 = {}, c10 = {}, c11 = {};
  for (int k = 0; k < DMODEL; k += 32) {
    v16h a0 = load_A(ap + k, DMODEL, lane);
    v16h a1 = load_A(ap + (size_t)16 * DMODEL + k, DMODEL, lane);
    v16h b0 = load_B(bp + k, DMODEL, lane);
    v16h b1 = load_B(bp + (size_t)16 * DMODEL + k, DMODEL, lane);
    c00 = wmma16(a0, b0, c00);
    c01 = wmma16(a0, b1, c01);
    c10 = wmma16(a1, b0, c10);
    c11 = wmma16(a1, b1, c11);
  }
  int ncol = lane & 15;
  int mb = (lane >> 4) << 3;

  auto store_tile = [&](v8f c, int mi, int ni) {
    int col = ct * 32 + ni * 16 + ncol;
    int h = col >> 6, d = col & 63;
#pragma unroll
    for (int i = 0; i < 8; ++i) {
      int m = mt * 32 + mi * 16 + mb + i;
      int b = m >> 12, n = m & (NSEQ - 1);
      float v = c[i];
      if (do_clip) v = fminf(5.0f, fmaxf(-5.0f, v));
      size_t idx = transpose_out
          ? (((size_t)(b * HEADS + h) * DHEAD + d) * NSEQ + n)
          : (((size_t)(b * HEADS + h) * NSEQ + n) * DHEAD + d);
      out[idx] = (half_t)v;
    }
  };
  store_tile(c00, 0, 0);
  store_tile(c01, 0, 1);
  store_tile(c10, 1, 0);
  store_tile(c11, 1, 1);
}

// ---------------------------------------------------------------------------
// Attention: one block per (b, h, 32-query tile). 8 waves, 256 KB LDS.
// Phase 1: S = scale * Q K^T, masked, f16 logits in LDS (32 x 4096); each
//          wave owns 512 keys and reuses each K B-fragment for 2 Q subtiles.
// Phase 2: row softmax in LDS (in-place, same type), 4 rows per wave.
// Phase 3: O = P V; 8 waves = 2 q-subtiles x 4 d-tiles, 128 WMMAs each.
__global__ void __launch_bounds__(256) attn_kernel(
    const half_t* __restrict__ qb, const half_t* __restrict__ kb,
    const half_t* __restrict__ vtb, const uint8_t* __restrict__ mask,
    half_t* __restrict__ ob) {
  extern __shared__ half_t S[];              // [32][4096] f16
  const int lane = threadIdx.x & 31;
  const int wave = threadIdx.x >> 5;
  const int qt = blockIdx.x & 127;           // 128 query tiles of 32
  const int bh = blockIdx.x >> 7;
  const int b  = bh >> 3;
  const int h  = bh & 7;
  const int ncol = lane & 15;
  const int mb   = (lane >> 4) << 3;

  // Q fragments for both 16-row subtiles (reused across all key tiles)
  const half_t* qp = qb + ((size_t)bh * NSEQ + (size_t)qt * 32) * DHEAD;
  v16h a00 = load_A(qp, DHEAD, lane);
  v16h a01 = load_A(qp + 32, DHEAD, lane);
  v16h a10 = load_A(qp + (size_t)16 * DHEAD, DHEAD, lane);
  v16h a11 = load_A(qp + (size_t)16 * DHEAD + 32, DHEAD, lane);

  const uint8_t* mrow = mask + (size_t)b * NSEQ;

  // ---- Phase 1: logits ----
  for (int kt = 0; kt < 32; ++kt) {
    int key0 = wave * 512 + kt * 16;
    const half_t* kp = kb + ((size_t)bh * NSEQ + key0) * DHEAD;
    v16h b0 = load_B(kp,      DHEAD, lane);
    v16h b1 = load_B(kp + 32, DHEAD, lane);
    v8f c0 = {};
    c0 = wmma16(a00, b0, c0);
    c0 = wmma16(a01, b1, c0);
    v8f c1 = {};
    c1 = wmma16(a10, b0, c1);
    c1 = wmma16(a11, b1, c1);
    int key = key0 + ncol;
    bool dead = (mrow[key] != 0);
#pragma unroll
    for (int i = 0; i < 8; ++i) {
      float v0 = dead ? -__builtin_huge_valf() : c0[i] * 0.125f;  // 64^-0.5
      float v1 = dead ? -__builtin_huge_valf() : c1[i] * 0.125f;
      S[(size_t)(mb + i) * NSEQ + key]        = (half_t)v0;
      S[(size_t)(16 + mb + i) * NSEQ + key]   = (half_t)v1;
    }
  }
  __syncthreads();

  // ---- Phase 2: softmax (4 rows per wave) ----
  for (int r = wave; r < 32; r += 8) {
    half_t* row = S + (size_t)r * NSEQ;
    float m = -__builtin_huge_valf();
    for (int j = lane; j < NSEQ; j += 32) m = fmaxf(m, (float)row[j]);
    m = wave_max(m);
    float s = 0.f;
    for (int j = lane; j < NSEQ; j += 32) {
      float e = __expf((float)row[j] - m);
      s += e;
      row[j] = (half_t)e;
    }
    s = wave_sum(s);
    float inv = 1.0f / s;
    for (int j = lane; j < NSEQ; j += 32)
      row[j] = (half_t)((float)row[j] * inv);
  }
  __syncthreads();

  // ---- Phase 3: O = P @ V (all 8 waves) ----
  {
    int qs = wave >> 2;                      // q subtile 0/1
    int dt = wave & 3;                       // d tile 0..3
    const half_t* vp = vtb + ((size_t)bh * DHEAD + (size_t)dt * 16) * NSEQ;
    const half_t* prow = S + (size_t)qs * 16 * NSEQ;
    v8f c = {};
    for (int kt = 0; kt < NSEQ / 32; ++kt) {
      v16h a  = load_A(prow + kt * 32, NSEQ, lane);    // probs from LDS
      v16h bb = load_B(vp + kt * 32, NSEQ, lane);      // d-major V
      c = wmma16(a, bb, c);
    }
    int col = h * DHEAD + dt * 16 + ncol;
    size_t rowbase = (size_t)b * NSEQ + (size_t)qt * 32 + qs * 16 + mb;
#pragma unroll
    for (int i = 0; i < 8; ++i)
      ob[(rowbase + i) * DMODEL + col] = (half_t)c[i];
  }
}

// ---------------------------------------------------------------------------
// Final projection: out = O @ W_out + b_out (fp32 output), 32x32 per wave.
__global__ void __launch_bounds__(256) out_gemm(const half_t* __restrict__ O,
    const half_t* __restrict__ Wt, const float* __restrict__ bias,
    float* __restrict__ out) {
  int lane = threadIdx.x & 31;
  int wave = threadIdx.x >> 5;
  int gw = blockIdx.x * 8 + wave;            // 4096 wave-tiles (32x32)
  int mt = gw >> 4;
  int ct = gw & 15;
  const half_t* ap = O + (size_t)mt * 32 * DMODEL;
  const half_t* bp = Wt + (size_t)ct * 32 * DMODEL;
  v8f c00 = {}, c01 = {}, c10 = {}, c11 = {};
  for (int k = 0; k < DMODEL; k += 32) {
    v16h a0 = load_A(ap + k, DMODEL, lane);
    v16h a1 = load_A(ap + (size_t)16 * DMODEL + k, DMODEL, lane);
    v16h b0 = load_B(bp + k, DMODEL, lane);
    v16h b1 = load_B(bp + (size_t)16 * DMODEL + k, DMODEL, lane);
    c00 = wmma16(a0, b0, c00);
    c01 = wmma16(a0, b1, c01);
    c10 = wmma16(a1, b0, c10);
    c11 = wmma16(a1, b1, c11);
  }
  int ncol = lane & 15;
  int mb = (lane >> 4) << 3;

  auto store_tile = [&](v8f c, int mi, int ni) {
    int col = ct * 32 + ni * 16 + ncol;
    float bv = bias[col];
#pragma unroll
    for (int i = 0; i < 8; ++i) {
      size_t m = (size_t)mt * 32 + mi * 16 + mb + i;
      out[m * DMODEL + col] = c[i] + bv;
    }
  };
  store_tile(c00, 0, 0);
  store_tile(c01, 0, 1);
  store_tile(c10, 1, 0);
  store_tile(c11, 1, 1);
}

// ---------------------------------------------------------------------------
extern "C" void kernel_launch(void* const* d_in, const int* in_sizes, int n_in,
                              void* d_out, int out_size, void* d_ws, size_t ws_size,
                              hipStream_t stream) {
  const float*   keys     = (const float*)d_in[0];
  const float*   values   = (const float*)d_in[1];
  const uint8_t* mask     = (const uint8_t*)d_in[2];   // jnp.bool_ -> bytes
  const float*   qk_gamma = (const float*)d_in[3];
  const float*   qk_beta  = (const float*)d_in[4];
  const float*   v_gamma  = (const float*)d_in[5];
  const float*   v_beta   = (const float*)d_in[6];
  const float*   W_qk     = (const float*)d_in[7];
  const float*   W_v      = (const float*)d_in[8];
  const float*   W_out    = (const float*)d_in[9];
  const float*   b_out    = (const float*)d_in[10];

  char* ws = (char*)d_ws;
  const size_t SZ_ACT = (size_t)MROWS * DMODEL * sizeof(half_t);   // 8 MB each
  half_t* kn    = (half_t*)(ws + 0 * SZ_ACT);
  half_t* vn    = (half_t*)(ws + 1 * SZ_ACT);
  half_t* qb    = (half_t*)(ws + 2 * SZ_ACT);
  half_t* kb    = (half_t*)(ws + 3 * SZ_ACT);
  half_t* vtb   = (half_t*)(ws + 4 * SZ_ACT);
  half_t* ob    = (half_t*)(ws + 5 * SZ_ACT);
  half_t* wqkt  = (half_t*)(ws + 6 * SZ_ACT);
  half_t* wvt   = (half_t*)(ws + 6 * SZ_ACT + (size_t)1024 * 512 * 2);
  half_t* woutt = (half_t*)(ws + 6 * SZ_ACT + (size_t)1024 * 512 * 2
                                            + (size_t)512 * 512 * 2);

  // Weight convert + transpose (f32 -> f16, N-major for B-fragments)
  wt_kernel<<<(512 * 1024 + 255) / 256, 256, 0, stream>>>(W_qk, wqkt, 512, 1024);
  wt_kernel<<<(512 * 512 + 255) / 256, 256, 0, stream>>>(W_v, wvt, 512, 512);
  wt_kernel<<<(512 * 512 + 255) / 256, 256, 0, stream>>>(W_out, woutt, 512, 512);

  // LayerNorms (fp32 math, f16 out)
  ln_kernel<<<MROWS / 8, 256, 0, stream>>>(keys, qk_gamma, qk_beta, kn);
  ln_kernel<<<MROWS / 8, 256, 0, stream>>>(values, v_gamma, v_beta, vn);

  // Projections: Q, K (clip, [b,h,N,64]), V (d-major [b,h,64,N])
  proj_kernel<<<512, 256, 0, stream>>>(kn, wqkt, qb, 1, 0);
  proj_kernel<<<512, 256, 0, stream>>>(kn, wqkt + (size_t)512 * 512, kb, 1, 0);
  proj_kernel<<<512, 256, 0, stream>>>(vn, wvt, vtb, 0, 1);

  // Attention: one block per (b,h,qtile32); 256 KB dynamic LDS
  attn_kernel<<<BATCH * HEADS * (NSEQ / 32), 256,
                (size_t)32 * NSEQ * sizeof(half_t), stream>>>(qb, kb, vtb, mask, ob);

  // Output projection + bias (fp32 out)
  out_gemm<<<512, 256, 0, stream>>>(ob, woutt, b_out, (float*)d_out);
}
